// RGCNRegressionModel_69518340653435
// MI455X (gfx1250) — matrix operand order repaired and checked by reading, not compiled, not run on hardware
//
#include <hip/hip_runtime.h>
#include <hip/hip_bf16.h>

typedef __attribute__((ext_vector_type(16))) __bf16 v16bf;
typedef __attribute__((ext_vector_type(8)))  float  v8f;

#define N_NODES  10000
#define N_EDGES  160000
#define N_REL    4
#define N_GRAPHS 256

union U2 { uint2 u; __bf16 h[4]; };

// ---------------------------------------------------------------- utilities

__global__ void zero_f32(float* __restrict__ p, size_t n) {
  size_t i = (size_t)blockIdx.x * blockDim.x + threadIdx.x;
  if (i < n) p[i] = 0.0f;
}

__global__ void cvt_f32_bf16(const float* __restrict__ in, __bf16* __restrict__ out,
                             size_t n) {
  size_t i = (size_t)blockIdx.x * blockDim.x + threadIdx.x;
  if (i < n) out[i] = (__bf16)in[i];
}

// per-(dst, relation) edge counts -> reciprocal
__global__ void count_edges(const int* __restrict__ dst, const int* __restrict__ et,
                            float* __restrict__ cnt) {
  int e = blockIdx.x * blockDim.x + threadIdx.x;
  if (e < N_EDGES) atomicAdd(&cnt[dst[e] * N_REL + et[e]], 1.0f);
}

__global__ void invert_cnt(float* __restrict__ cnt, int n) {
  int i = blockIdx.x * blockDim.x + threadIdx.x;
  if (i < n) cnt[i] = 1.0f / fmaxf(cnt[i], 1.0f);
}

// scatter-add x[src] into agg[dst][rel][:] (f32 accumulation, node-major layout)
__global__ void aggregate_edges(const int* __restrict__ src, const int* __restrict__ dst,
                                const int* __restrict__ et, const __bf16* __restrict__ x,
                                float* __restrict__ agg, int Dsh) {
  int e  = blockIdx.x;
  int s  = src[e];
  int nd = dst[e];
  int r  = et[e];
  int Dh = 1 << (Dsh - 1);
  float* out = agg + (((size_t)nd * N_REL + r) << Dsh);
  const __bf16* xin = x + ((size_t)s << Dsh);
  for (int d = threadIdx.x; d < Dh; d += blockDim.x) {
    __bf16 a = xin[2 * d];
    __bf16 b = xin[2 * d + 1];
    atomicAdd(&out[2 * d],     (float)a);
    atomicAdd(&out[2 * d + 1], (float)b);
  }
}

// Build packed bf16 A matrix: Apack[n][k] = k<RD ? bf16(agg[n*RD+k] * inv[n,k>>Dsh])
//                                                 : x[n][k-RD]
__global__ __launch_bounds__(256)
void build_apack(const float* __restrict__ agg, const float* __restrict__ inv,
                 const __bf16* __restrict__ x, __bf16* __restrict__ Apack,
                 int K, int RD, int Dsh) {
  int n = blockIdx.x;
  const float*  arow = agg + (size_t)n * RD;
  const __bf16* xrow = x + (size_t)n * (K - RD);
  __bf16*       orow = Apack + (size_t)n * K;
  const float*  iv   = inv + (size_t)n * N_REL;
  for (int k = threadIdx.x * 4; k < RD; k += blockDim.x * 4) {
    float  s = iv[k >> Dsh];
    float4 f = *(const float4*)(arow + k);
    U2 o;
    o.h[0] = (__bf16)(f.x * s); o.h[1] = (__bf16)(f.y * s);
    o.h[2] = (__bf16)(f.z * s); o.h[3] = (__bf16)(f.w * s);
    *(uint2*)(orow + k) = o.u;
  }
  int D = K - RD;
  for (int k = threadIdx.x * 4; k < D; k += blockDim.x * 4)
    *(uint2*)(orow + RD + k) = *(const uint2*)(xrow + k);
}

// pack transposed B: BpT[f*K + k] = bf16( k<RD ? W[k,f] : root[k-RD,f] )
// 32x32 LDS-tiled transpose; grid (K/32, ceil(F/32)), 256 threads (32x8).
__global__ __launch_bounds__(256)
void pack_b_t(const float* __restrict__ W, const float* __restrict__ root,
              __bf16* __restrict__ BpT, int RD, int K, int F) {
  __shared__ float T[32][33];
  const int k0 = blockIdx.x * 32;
  const int f0 = blockIdx.y * 32;
  const int tx = threadIdx.x & 31;
  const int ty = threadIdx.x >> 5;      // 0..7
#pragma unroll
  for (int j = 0; j < 4; ++j) {
    int k = k0 + ty + j * 8;
    int f = f0 + tx;
    float v = 0.0f;
    if (f < F) v = (k < RD) ? W[(size_t)k * F + f] : root[(size_t)(k - RD) * F + f];
    T[ty + j * 8][tx] = v;
  }
  __syncthreads();
#pragma unroll
  for (int j = 0; j < 4; ++j) {
    int f = f0 + ty + j * 8;
    int k = k0 + tx;
    if (f < F) BpT[(size_t)f * K + k] = (__bf16)T[tx][ty + j * 8];
  }
}

// ---------------------------------------------------------------- WMMA GEMM
// out[n,f] = relu( A (N x K, bf16) @ B (K x F) + bias ), B given transposed [F][K].
// Block: 256 thr (8 waves, 2x4), tile 64(M) x 128(N), K-step 64 (2 x 32 sub-steps),
// reg-pipelined global->LDS staging (named scalars, no address-taken locals).
// K must be a multiple of 64. Out-of-range rows/cols CLAMPED at load (masked at store).
__global__ __launch_bounds__(256)
void gemm_bias_relu(const __bf16* __restrict__ A, const __bf16* __restrict__ Bt,
                    const float* __restrict__ bias, __bf16* __restrict__ out,
                    int K, int F) {
  __shared__ __align__(16) __bf16 As[64 * 72];    // [m][k], stride 72 halves
  __shared__ __align__(16) __bf16 Bs[128 * 72];   // transposed [n][k], stride 72 halves

  const int tid  = threadIdx.x;
  const int lane = tid & 31;
  const int wid  = tid >> 5;
  const int wm   = wid >> 2;          // 0..1
  const int wn   = wid & 3;           // 0..3
  const int hg   = lane >> 4;         // half-group (lane>=16)
  const int mr   = lane & 15;
  const int rowBase = blockIdx.y * 64;
  const int colBase = blockIdx.x * 128;

  // --- per-thread staging geometry (constant across K loop) ---
  const int an  = tid >> 2;                 // A row 0..63
  const int akc = (tid & 3) * 16;           // A k-chunk 0/16/32/48
  const int agn = min(rowBase + an, N_NODES - 1);          // clamp, masked at store
  uint4* const asp0 = (uint4*)&As[an * 72 + akc];
  uint4* const asp1 = (uint4*)&As[an * 72 + akc + 8];
  const __bf16* const arow = A + (size_t)agn * K + akc;

  const int bn  = tid >> 1;                 // B col 0..127
  const int bkc = (tid & 1) * 32;           // B k-chunk 0/32
  const int gf  = min(colBase + bn, F - 1);                // clamp, masked at store
  uint4* const bsp0 = (uint4*)&Bs[bn * 72 + bkc];
  uint4* const bsp1 = (uint4*)&Bs[bn * 72 + bkc + 8];
  uint4* const bsp2 = (uint4*)&Bs[bn * 72 + bkc + 16];
  uint4* const bsp3 = (uint4*)&Bs[bn * 72 + bkc + 24];
  const __bf16* const brow = Bt + (size_t)gf * K + bkc;

  // --- fragment LDS base pointers (constant across K loop) ---
  const __bf16* const pa0 = &As[(wm * 32 + mr) * 72];
  const __bf16* const pa1 = &As[(wm * 32 + 16 + mr) * 72];
  const __bf16* const pb0 = &Bs[(wn * 32 + mr) * 72 + hg * 16];
  const __bf16* const pb1 = &Bs[(wn * 32 + 16 + mr) * 72 + hg * 16];

  const v8f vzero = {0.f, 0.f, 0.f, 0.f, 0.f, 0.f, 0.f, 0.f};
  v8f acc[2][2];
  acc[0][0] = vzero; acc[0][1] = vzero; acc[1][0] = vzero; acc[1][1] = vzero;

  // prologue stage (k0 = 0)
  uint4 ra0 = *(const uint4*)(arow);
  uint4 ra1 = *(const uint4*)(arow + 8);
  uint4 rb0 = *(const uint4*)(brow);
  uint4 rb1 = *(const uint4*)(brow + 8);
  uint4 rb2 = *(const uint4*)(brow + 16);
  uint4 rb3 = *(const uint4*)(brow + 24);

  for (int k0 = 0; k0 < K; k0 += 64) {
    *asp0 = ra0; *asp1 = ra1;
    *bsp0 = rb0; *bsp1 = rb1; *bsp2 = rb2; *bsp3 = rb3;
    __syncthreads();

    const int kn = k0 + 64;
    if (kn < K) {                          // stage next tile, overlap with compute
      ra0 = *(const uint4*)(arow + kn);
      ra1 = *(const uint4*)(arow + kn + 8);
      rb0 = *(const uint4*)(brow + kn);
      rb1 = *(const uint4*)(brow + kn + 8);
      rb2 = *(const uint4*)(brow + kn + 16);
      rb3 = *(const uint4*)(brow + kn + 24);
    }

#pragma unroll
    for (int s = 0; s < 2; ++s) {
      const int ko = s * 32;
      // fragments: contiguous k order (k = 16*hg + s) for both A and B
      v16bf afr[2], bfr[2];
      ((uint4*)&afr[0])[0] = *(const uint4*)(pa0 + ko + hg * 16);
      ((uint4*)&afr[0])[1] = *(const uint4*)(pa0 + ko + hg * 16 + 8);
      ((uint4*)&afr[1])[0] = *(const uint4*)(pa1 + ko + hg * 16);
      ((uint4*)&afr[1])[1] = *(const uint4*)(pa1 + ko + hg * 16 + 8);
      ((uint4*)&bfr[0])[0] = *(const uint4*)(pb0 + ko);
      ((uint4*)&bfr[0])[1] = *(const uint4*)(pb0 + ko + 8);
      ((uint4*)&bfr[1])[0] = *(const uint4*)(pb1 + ko);
      ((uint4*)&bfr[1])[1] = *(const uint4*)(pb1 + ko + 8);

#pragma unroll
      for (int im = 0; im < 2; ++im)
#pragma unroll
        for (int in = 0; in < 2; ++in)
          acc[im][in] = __builtin_amdgcn_wmma_f32_16x16x32_bf16(
              false, afr[im], false, bfr[in], (short)0, acc[im][in], false, false);
    }

    __syncthreads();
  }

  // epilogue: bias + relu + bf16 store (C layout: row = v + 8*hg, col = mr)
#pragma unroll
  for (int im = 0; im < 2; ++im) {
#pragma unroll
    for (int in = 0; in < 2; ++in) {
      int gc = colBase + wn * 32 + in * 16 + mr;
      if (gc >= F) continue;
      float b = bias[gc];
      int gr0 = rowBase + wm * 32 + im * 16 + hg * 8;
#pragma unroll
      for (int v = 0; v < 8; ++v) {
        int gr = gr0 + v;
        if (gr < N_NODES) {
          float val = fmaxf(acc[im][in][v] + b, 0.0f);
          out[(size_t)gr * F + gc] = (__bf16)val;
        }
      }
    }
  }
}

// ---------------------------------------------------------------- pooling + MLP

__global__ void pool_scatter(const __bf16* __restrict__ h, const int* __restrict__ batch,
                             float* __restrict__ pool, float* __restrict__ gcnt, int F) {
  int n = blockIdx.x;
  int b = batch[n];
  const __bf16* hp = h + (size_t)n * F;
  float* pp = pool + (size_t)b * F;
  for (int f = threadIdx.x; f < F; f += blockDim.x)
    atomicAdd(&pp[f], (float)hp[f]);
  if (threadIdx.x == 0) atomicAdd(&gcnt[b], 1.0f);
}

__global__ void mlp_head(const float* __restrict__ pool, const float* __restrict__ gcnt,
                         const float* __restrict__ fw1, const float* __restrict__ fb1,
                         const float* __restrict__ fw2, const float* __restrict__ fb2,
                         float* __restrict__ out) {
  int g = blockIdx.x;
  int j = threadIdx.x;  // 0..63
  float invc = 1.0f / fmaxf(gcnt[g], 1.0f);
  const float* p = pool + (size_t)g * 712;
  float s = fb1[j];
  for (int k = 0; k < 712; ++k)
    s = fmaf(p[k] * invc, fw1[k * 64 + j], s);
  s = fmaxf(s, 0.0f);
  __shared__ float red[64];
  red[j] = s * fw2[j];
  __syncthreads();
  if (j == 0) {
    float t = 0.0f;
    for (int k = 0; k < 64; ++k) t += red[k];
    out[g] = t + fb2[0];
  }
}

// ---------------------------------------------------------------- launch

extern "C" void kernel_launch(void* const* d_in, const int* in_sizes, int n_in,
                              void* d_out, int out_size, void* d_ws, size_t ws_size,
                              hipStream_t stream) {
  (void)in_sizes; (void)n_in; (void)out_size; (void)ws_size;

  const float* x     = (const float*)d_in[0];
  const int*   ei    = (const int*)d_in[1];
  const int*   et    = (const int*)d_in[2];
  const int*   batch = (const int*)d_in[3];
  const float* W1    = (const float*)d_in[4];
  const float* root1 = (const float*)d_in[5];
  const float* b1    = (const float*)d_in[6];
  const float* W2    = (const float*)d_in[7];
  const float* root2 = (const float*)d_in[8];
  const float* b2    = (const float*)d_in[9];
  const float* W3    = (const float*)d_in[10];
  const float* root3 = (const float*)d_in[11];
  const float* b3    = (const float*)d_in[12];
  const float* fw1   = (const float*)d_in[13];
  const float* fb1   = (const float*)d_in[14];
  const float* fw2   = (const float*)d_in[15];
  const float* fb2   = (const float*)d_in[16];
  const int* src = ei;
  const int* dst = ei + N_EDGES;

  // workspace carve-out
  char* ws = (char*)d_ws;
  size_t off = 0;
  auto take = [&](size_t bytes) -> char* {
    char* p = ws + off;
    off += (bytes + 255) & ~(size_t)255;
    return p;
  };
  float*  inv   = (float*)take((size_t)N_NODES * N_REL * 4);
  float*  agg   = (float*)take((size_t)N_NODES * N_REL * 1024 * 4);   // 164 MB, [n][r][d]
  __bf16* Apack = (__bf16*)take((size_t)N_NODES * 5120 * 2);          // 102 MB
  __bf16* BpT   = (__bf16*)take((size_t)1024 * 5120 * 2);             // 10.5 MB
  __bf16* hA    = (__bf16*)take((size_t)N_NODES * 1024 * 2);
  __bf16* hB    = (__bf16*)take((size_t)N_NODES * 1024 * 2);
  float*  pool  = (float*)take((size_t)N_GRAPHS * 712 * 4);
  float*  gcnt  = (float*)take((size_t)N_GRAPHS * 4);

  auto cdiv = [](size_t a, size_t b) { return (unsigned)((a + b - 1) / b); };

  // edge counts (shared by all layers)
  zero_f32<<<cdiv((size_t)N_NODES * N_REL, 256), 256, 0, stream>>>(inv, (size_t)N_NODES * N_REL);
  count_edges<<<cdiv(N_EDGES, 256), 256, 0, stream>>>(dst, et, inv);
  invert_cnt<<<cdiv((size_t)N_NODES * N_REL, 256), 256, 0, stream>>>(inv, N_NODES * N_REL);

  // x -> bf16 (stored in hA as [N,64])
  cvt_f32_bf16<<<cdiv((size_t)N_NODES * 64, 256), 256, 0, stream>>>(x, hA, (size_t)N_NODES * 64);

  // ---- layer 1: D=64, K=320, F=1024, in=hA, out=hB
  {
    const int Dsh = 6, K = 320, F = 1024, RD = 256;
    zero_f32<<<cdiv((size_t)N_NODES * RD, 256), 256, 0, stream>>>(agg, (size_t)N_NODES * RD);
    aggregate_edges<<<N_EDGES, 32, 0, stream>>>(src, dst, et, hA, agg, Dsh);
    build_apack<<<N_NODES, 256, 0, stream>>>(agg, inv, hA, Apack, K, RD, Dsh);
    pack_b_t<<<dim3(K / 32, cdiv(F, 32)), 256, 0, stream>>>(W1, root1, BpT, RD, K, F);
    gemm_bias_relu<<<dim3(cdiv(F, 128), cdiv(N_NODES, 64)), 256, 0, stream>>>(
        Apack, BpT, b1, hB, K, F);
  }
  // ---- layer 2: D=1024, K=5120, F=1024, in=hB, out=hA
  {
    const int Dsh = 10, K = 5120, F = 1024, RD = 4096;
    zero_f32<<<cdiv((size_t)N_NODES * RD, 256), 256, 0, stream>>>(agg, (size_t)N_NODES * RD);
    aggregate_edges<<<N_EDGES, 256, 0, stream>>>(src, dst, et, hB, agg, Dsh);
    build_apack<<<N_NODES, 256, 0, stream>>>(agg, inv, hB, Apack, K, RD, Dsh);
    pack_b_t<<<dim3(K / 32, cdiv(F, 32)), 256, 0, stream>>>(W2, root2, BpT, RD, K, F);
    gemm_bias_relu<<<dim3(cdiv(F, 128), cdiv(N_NODES, 64)), 256, 0, stream>>>(
        Apack, BpT, b2, hA, K, F);
  }
  // ---- layer 3: D=1024, K=5120, F=712, in=hA, out=hB
  {
    const int Dsh = 10, K = 5120, F = 712, RD = 4096;
    zero_f32<<<cdiv((size_t)N_NODES * RD, 256), 256, 0, stream>>>(agg, (size_t)N_NODES * RD);
    aggregate_edges<<<N_EDGES, 256, 0, stream>>>(src, dst, et, hA, agg, Dsh);
    build_apack<<<N_NODES, 256, 0, stream>>>(agg, inv, hA, Apack, K, RD, Dsh);
    pack_b_t<<<dim3(K / 32, cdiv(F, 32)), 256, 0, stream>>>(W3, root3, BpT, RD, K, F);
    gemm_bias_relu<<<dim3(cdiv(F, 128), cdiv(N_NODES, 64)), 256, 0, stream>>>(
        Apack, BpT, b3, hB, K, F);
  }

  // ---- global mean pool + MLP head
  zero_f32<<<cdiv((size_t)N_GRAPHS * 712, 256), 256, 0, stream>>>(pool, (size_t)N_GRAPHS * 712);
  zero_f32<<<1, 256, 0, stream>>>(gcnt, N_GRAPHS);
  pool_scatter<<<N_NODES, 256, 0, stream>>>(hB, batch, pool, gcnt, 712);
  mlp_head<<<N_GRAPHS, 64, 0, stream>>>(pool, gcnt, fw1, fb1, fw2, fb2, (float*)d_out);
}
